// TriangularTransformerNodeRegressor_51900384805521
// MI455X (gfx1250) — matrix operand order repaired and checked by reading, not compiled, not run on hardware
//
#include <hip/hip_runtime.h>

// ---------------------------------------------------------------------------
// Types for CDNA5 WMMA (wave32): 16x16x32 bf16 -> f32 accumulate
// ---------------------------------------------------------------------------
typedef __attribute__((ext_vector_type(16))) __bf16 v16bf;
typedef __attribute__((ext_vector_type(8)))  float  v8f;

union AFrag { v16bf v; unsigned int u[8]; };

__device__ __forceinline__ unsigned short f2bf(float f) {
  union { float f; unsigned int u; } x; x.f = f;
  unsigned int r = x.u + 0x7FFFu + ((x.u >> 16) & 1u);   // round-to-nearest-even
  return (unsigned short)(r >> 16);
}
__device__ __forceinline__ float bflo(unsigned int w) {
  union { unsigned int u; float f; } x; x.u = w << 16; return x.f;
}
__device__ __forceinline__ float bfhi(unsigned int w) {
  union { unsigned int u; float f; } x; x.u = w & 0xFFFF0000u; return x.f;
}

// ---------------------------------------------------------------------------
// Blocked bf16 WMMA GEMM with async-to-LDS weight staging.
//   C[M x Nn] = A[M x K] * B[K x Nn] (+bias, optional relu)
//   A row-major (lda); BT = B^T row-major N x K (ldb).
// Block = 256 threads = 8 waves; block tile = 128(M) x 64(N).
//   wave w -> M-tile (bm*8+w), N-tiles bn*4 .. bn*4+3 (16x64 per wave).
// B slab (64 n-rows x 32 k, bf16 = 4KB) is shared by all 8 waves: each thread
// async-copies one 16B chunk global->LDS (GLOBAL_LOAD_ASYNC_TO_LDS_B128,
// tracked by ASYNCcnt), double buffered; one s_wait_asynccnt + barrier per
// K-step.  A fragments stream from global (unique per wave).
// Fragment layouts used here (wave32; A/B contiguous k, not the ISA 7.12.2 table order):
//   A 16-bit 16x32: lane m=lane&15, half=lane>>4; vgpr v holds
//                   K = half*16 + 2v (+1)
//   B 16-bit 32x16: lane n=lane&15, half=lane>>4; vgpr v holds K = half*16+2v (+1)
//     -> 16 consecutive ushorts per lane in LDS => ds_load_b128 pairs
//   C f32 16x16   : row = (lane>>4)*8 + v, col = lane&15
// ---------------------------------------------------------------------------
__global__ void __launch_bounds__(256)
gemm_bf16_wmma_lds(const unsigned short* __restrict__ A, int lda,
                   const unsigned short* __restrict__ BT, int ldb,
                   float* __restrict__ Cf, unsigned short* __restrict__ Cb, int ldc,
                   const float* __restrict__ bias, int M, int Nn, int K, int relu)
{
  __shared__ __align__(16) unsigned short shB[2][64 * 32];

  const int t    = threadIdx.x;
  const int lane = t & 31;
  const int w    = t >> 5;                 // wave 0..7
  const int nblk = Nn >> 6;
  const int bm   = blockIdx.x / nblk;
  const int bn   = blockIdx.x - bm * nblk;
  const int l15  = lane & 15;
  const int half = lane >> 4;

  const int mrow = bm * 128 + w * 16 + l15;
  const unsigned short* Ap = A + (size_t)mrow * lda;

  // Each thread owns one 16B chunk of the 64x32 slab: row t>>2, k-chunk t&3.
  const unsigned short* Bg = BT + (size_t)(bn * 64 + (t >> 2)) * ldb + (t & 3) * 8;
  const int dstIdx = (t >> 2) * 32 + (t & 3) * 8;

  const int iters = K >> 5;
  {
    const unsigned lds0 = (unsigned)(uintptr_t)&shB[0][dstIdx];
    asm volatile("global_load_async_to_lds_b128 %0, %1, off"
                 :: "v"(lds0), "v"((unsigned long long)(uintptr_t)Bg)
                 : "memory");
  }

  v8f acc[4] = {v8f{}, v8f{}, v8f{}, v8f{}};

  for (int it = 0; it < iters; ++it) {
    // my async chunk (and, transitively via the barrier, everyone's) is done
    asm volatile("s_wait_asynccnt 0x0" ::: "memory");
    __syncthreads();
    if (it + 1 < iters) {   // prefetch next slab into the other buffer
      const unsigned ldsn = (unsigned)(uintptr_t)&shB[(it + 1) & 1][dstIdx];
      asm volatile("global_load_async_to_lds_b128 %0, %1, off"
                   :: "v"(ldsn),
                      "v"((unsigned long long)(uintptr_t)(Bg + (size_t)(it + 1) * 32))
                   : "memory");
    }
    const int k0 = it << 5;
    AFrag a;
#pragma unroll
    for (int v = 0; v < 8; ++v) {
      const int ka = (half << 4) + (v << 1);
      a.u[v] = *(const unsigned int*)(Ap + k0 + ka);
    }
    const unsigned short* sb = &shB[it & 1][0];
    AFrag b[4];
#pragma unroll
    for (int t4 = 0; t4 < 4; ++t4) {
      const unsigned short* bp = sb + (t4 * 16 + l15) * 32 + (half << 4);
#pragma unroll
      for (int v = 0; v < 8; ++v) b[t4].u[v] = ((const unsigned int*)bp)[v];
    }
    acc[0] = __builtin_amdgcn_wmma_f32_16x16x32_bf16(false, a.v, false, b[0].v,
                                                     (short)0, acc[0], false, false);
    acc[1] = __builtin_amdgcn_wmma_f32_16x16x32_bf16(false, a.v, false, b[1].v,
                                                     (short)0, acc[1], false, false);
    acc[2] = __builtin_amdgcn_wmma_f32_16x16x32_bf16(false, a.v, false, b[2].v,
                                                     (short)0, acc[2], false, false);
    acc[3] = __builtin_amdgcn_wmma_f32_16x16x32_bf16(false, a.v, false, b[3].v,
                                                     (short)0, acc[3], false, false);
  }

#pragma unroll
  for (int t4 = 0; t4 < 4; ++t4) {
    const int col = bn * 64 + t4 * 16 + l15;
    const float bv = bias ? bias[col] : 0.0f;
#pragma unroll
    for (int v = 0; v < 8; ++v) {
      const int row = bm * 128 + w * 16 + (half << 3) + v;
      float y = acc[t4][v] + bv;
      if (relu) y = fmaxf(y, 0.0f);
      if (Cf) Cf[(size_t)row * ldc + col] = y;
      if (Cb) Cb[(size_t)row * ldc + col] = f2bf(y);
    }
  }
}

// ---------------------------------------------------------------------------
// Attention scores: for fixed (b,h,l): S[i,j] = (Q_i[l,:] . K[l,j,:]) / sqrt(DH)
// Q/K/V1/V2 fused in qkvv[m][512] (cols 0/128/256/384 + h*32).
// One wave per 16x16 (i,j) tile; 16 tiles per (b,h,l); single K=32 WMMA.
// ---------------------------------------------------------------------------
__global__ void __launch_bounds__(256)
attn_scores_wmma(const unsigned short* __restrict__ qkvv, float* __restrict__ s)
{
  const int lane = threadIdx.x & 31;
  const int wave = (blockIdx.x * blockDim.x + threadIdx.x) >> 5;
  if (wave >= 8 * 4 * 64 * 16) return;
  const int tile = wave & 15;
  const int ti = tile >> 2, tj = tile & 3;
  const int rest = wave >> 4;
  const int l = rest & 63;
  const int bh = rest >> 6;
  const int h = bh & 3, b = bh >> 2;
  const int l15 = lane & 15;
  const int half = lane >> 4;

  const unsigned short* qp =
      qkvv + ((size_t)((b * 64 + ti * 16 + l15) * 64 + l)) * 512 + h * 32;
  const unsigned short* kp =
      qkvv + ((size_t)((b * 64 + l) * 64 + tj * 16 + l15)) * 512 + 128 + h * 32;

  AFrag a, bb;
#pragma unroll
  for (int v = 0; v < 8; ++v) {
    const int ka = (half << 4) + (v << 1);
    const int kb = (half << 4) + (v << 1);
    a.u[v]  = *(const unsigned int*)(qp + ka);
    bb.u[v] = *(const unsigned int*)(kp + kb);
  }
  v8f acc = {};
  acc = __builtin_amdgcn_wmma_f32_16x16x32_bf16(false, a.v, false, bb.v,
                                                (short)0, acc, false, false);
  const float inv_scale = 0.17677669529663687f;  // 1/sqrt(32)
  const int j = tj * 16 + l15;
#pragma unroll
  for (int v = 0; v < 8; ++v) {
    const int i = ti * 16 + (half << 3) + v;
    s[(((size_t)(b * 4 + h) * 64 + i) * 64 + l) * 64 + j] = acc[v] * inv_scale;
  }
}

// Softmax over l for each (b,h,i,j); element stride between l's is 64 floats.
__global__ void __launch_bounds__(256)
softmax_l(float* __restrict__ s)
{
  const int t = blockIdx.x * blockDim.x + threadIdx.x;
  if (t >= 8 * 4 * 64 * 64) return;
  const int j = t & 63;
  const int i = (t >> 6) & 63;
  const int bh = t >> 12;
  float* p = s + (((size_t)bh * 64 + i) * 64) * 64 + j;
  float mx = -1e30f;
  for (int l = 0; l < 64; ++l) mx = fmaxf(mx, p[l * 64]);
  float sum = 0.0f;
  for (int l = 0; l < 64; ++l) { float e = __expf(p[l * 64] - mx); p[l * 64] = e; sum += e; }
  const float r = 1.0f / sum;
  for (int l = 0; l < 64; ++l) p[l * 64] *= r;
}

// o[b,i,j,h,d] = sum_l a[b,h,i,l,j] * v1[b,i,l,h,d] * v2[b,l,j,h,d]
// Gated contraction (not a GEMM) -> VALU; thread handles 8 d-values.
__global__ void __launch_bounds__(256)
tri_o_einsum(const float* __restrict__ s, const unsigned short* __restrict__ qkvv,
             unsigned short* __restrict__ oB)
{
  const int t = blockIdx.x * blockDim.x + threadIdx.x;
  if (t >= 8 * 4 * 64 * 64 * 4) return;
  const int dg = t & 3;
  const int j = (t >> 2) & 63;
  const int i = (t >> 8) & 63;
  const int h = (t >> 14) & 3;
  const int b = t >> 16;
  const float* ap = s + (((size_t)(b * 4 + h) * 64 + i) * 64) * 64 + j;
  float acc[8] = {0, 0, 0, 0, 0, 0, 0, 0};
  for (int l = 0; l < 64; ++l) {
    const float a = ap[l * 64];
    const uint4 w1 = *(const uint4*)(qkvv +
        ((size_t)((b * 64 + i) * 64 + l)) * 512 + 256 + h * 32 + dg * 8);
    const uint4 w2 = *(const uint4*)(qkvv +
        ((size_t)((b * 64 + l) * 64 + j)) * 512 + 384 + h * 32 + dg * 8);
    const unsigned int u1[4] = {w1.x, w1.y, w1.z, w1.w};
    const unsigned int u2[4] = {w2.x, w2.y, w2.z, w2.w};
#pragma unroll
    for (int q = 0; q < 4; ++q) {
      acc[2 * q]     += a * bflo(u1[q]) * bflo(u2[q]);
      acc[2 * q + 1] += a * bfhi(u1[q]) * bfhi(u2[q]);
    }
  }
  unsigned short* op = oB + ((size_t)((b * 64 + i) * 64 + j)) * 128 + h * 32 + dg * 8;
#pragma unroll
  for (int q = 0; q < 8; ++q) op[q] = f2bf(acc[q]);
}

// tok = LN(tok + add) * g + b; one wave32 per token row of 128.
__global__ void __launch_bounds__(256)
resid_layernorm(float* __restrict__ tokF, unsigned short* __restrict__ tokB,
                const float* __restrict__ add, const float* __restrict__ g,
                const float* __restrict__ be)
{
  const int lane = threadIdx.x & 31;
  const int m = (blockIdx.x * blockDim.x + threadIdx.x) >> 5;
  if (m >= 32768) return;
  float* tp = tokF + (size_t)m * 128;
  const float* ap = add + (size_t)m * 128;
  unsigned short* bp = tokB + (size_t)m * 128;
  float x[4];
  float sum = 0.0f;
#pragma unroll
  for (int q = 0; q < 4; ++q) { const int c = q * 32 + lane; x[q] = tp[c] + ap[c]; sum += x[q]; }
#pragma unroll
  for (int o = 16; o > 0; o >>= 1) sum += __shfl_xor(sum, o, 32);
  const float mean = sum * (1.0f / 128.0f);
  float var = 0.0f;
#pragma unroll
  for (int q = 0; q < 4; ++q) { const float d = x[q] - mean; var += d * d; }
#pragma unroll
  for (int o = 16; o > 0; o >>= 1) var += __shfl_xor(var, o, 32);
  const float rs = rsqrtf(var * (1.0f / 128.0f) + 1e-5f);
#pragma unroll
  for (int q = 0; q < 4; ++q) {
    const int c = q * 32 + lane;
    const float y = (x[q] - mean) * rs * g[c] + be[c];
    tp[c] = y;
    bp[c] = f2bf(y);
  }
}

// tokF currently holds edge_emb; apply no_edge / diagonal node_emb, emit bf16.
__global__ void __launch_bounds__(256)
mask_embed(float* __restrict__ tokF, unsigned short* __restrict__ tokB,
           const float* __restrict__ nodeEmb, const float* __restrict__ no_edge,
           const int* __restrict__ mask)
{
  const int t = blockIdx.x * blockDim.x + threadIdx.x;
  if (t >= 32768 * 128) return;
  const int c = t & 127;
  const int m = t >> 7;
  const int j = m & 63;
  const int i = (m >> 6) & 63;
  const int b = m >> 12;
  float v = tokF[t];
  if (i == j)            v = nodeEmb[(size_t)(b * 64 + i) * 128 + c];
  else if (mask[m] == 0) v = no_edge[c];
  tokF[t] = v;
  tokB[t] = f2bf(v);
}

// src f32 [Mr x K] -> dst bf16 [Mr x KP], zero-padded in K.
__global__ void __launch_bounds__(256)
conv_pad_bf16(const float* __restrict__ src, unsigned short* __restrict__ dst,
              int Mr, int K, int KP)
{
  const int t = blockIdx.x * blockDim.x + threadIdx.x;
  if (t >= Mr * KP) return;
  const int kp = t % KP;
  const int m = t / KP;
  dst[t] = (kp < K) ? f2bf(src[(size_t)m * K + kp]) : (unsigned short)0;
}

// src f32 [K x Nn] -> dst bf16 [Nn x KP] (= B^T, zero-padded in K).
__global__ void __launch_bounds__(256)
wconv_transpose_bf16(const float* __restrict__ src, unsigned short* __restrict__ dst,
                     int K, int Nn, int KP)
{
  const int t = blockIdx.x * blockDim.x + threadIdx.x;
  if (t >= Nn * KP) return;
  const int kp = t % KP;
  const int n = t / KP;
  dst[t] = (kp < K) ? f2bf(src[(size_t)kp * Nn + n]) : (unsigned short)0;
}

// out[b,i] = diag_tok[b,i,:] . Wout + bout ; one wave per (b,i).
__global__ void __launch_bounds__(256)
head_out_k(const float* __restrict__ tokF, const float* __restrict__ Wout,
           const float* __restrict__ bout, float* __restrict__ out)
{
  const int lane = threadIdx.x & 31;
  const int m = (blockIdx.x * blockDim.x + threadIdx.x) >> 5;
  if (m >= 512) return;
  const int b = m >> 6, i = m & 63;
  const float* tp = tokF + ((size_t)((b * 64 + i) * 64 + i)) * 128;
  float sum = 0.0f;
#pragma unroll
  for (int q = 0; q < 4; ++q) { const int c = q * 32 + lane; sum += tp[c] * Wout[c]; }
#pragma unroll
  for (int o = 16; o > 0; o >>= 1) sum += __shfl_xor(sum, o, 32);
  if (lane == 0) out[m] = sum + bout[0];
}

// ---------------------------------------------------------------------------
extern "C" void kernel_launch(void* const* d_in, const int* in_sizes, int n_in,
                              void* d_out, int out_size, void* d_ws, size_t ws_size,
                              hipStream_t stream)
{
  (void)in_sizes; (void)n_in; (void)out_size; (void)ws_size;
  const float* x         = (const float*)d_in[0];
  const float* edge_attr = (const float*)d_in[1];
  const int*   edge_mask = (const int*)d_in[2];
  const float* node_W = (const float*)d_in[3];
  const float* node_b = (const float*)d_in[4];
  const float* edge_W = (const float*)d_in[5];
  const float* edge_b = (const float*)d_in[6];
  const float* no_edge= (const float*)d_in[7];
  const float* Wq  = (const float*)d_in[8];
  const float* Wk  = (const float*)d_in[9];
  const float* Wv1 = (const float*)d_in[10];
  const float* Wv2 = (const float*)d_in[11];
  const float* Wo  = (const float*)d_in[12];
  const float* bo  = (const float*)d_in[13];
  const float* ln1g= (const float*)d_in[14];
  const float* ln1b= (const float*)d_in[15];
  const float* W1  = (const float*)d_in[16];
  const float* b1  = (const float*)d_in[17];
  const float* W2  = (const float*)d_in[18];
  const float* b2  = (const float*)d_in[19];
  const float* ln2g= (const float*)d_in[20];
  const float* ln2b= (const float*)d_in[21];
  const float* Wout= (const float*)d_in[22];
  const float* bout= (const float*)d_in[23];
  float* out = (float*)d_out;

  const int Bb = 8, Nn = 64, D = 128, H = 4, FF = 512, L = 3;
  const int M = Bb * Nn * Nn;  // 32768 tokens

  char* ws = (char*)d_ws;
  size_t off = 0;
  auto take = [&](size_t bytes) -> char* {
    char* p = ws + off;
    off += (bytes + 255) & ~(size_t)255;
    return p;
  };
  float*          tokF   = (float*)          take((size_t)M * D * 4);
  unsigned short* tokB   = (unsigned short*) take((size_t)M * D * 2);
  float*          nodeEmb= (float*)          take((size_t)Bb * Nn * D * 4);
  unsigned short* Ax     = (unsigned short*) take((size_t)Bb * Nn * 32 * 2);
  unsigned short* Ae     = (unsigned short*) take((size_t)M * 32 * 2);
  unsigned short* nodeWT = (unsigned short*) take((size_t)D * 32 * 2);
  unsigned short* edgeWT = (unsigned short*) take((size_t)D * 32 * 2);
  unsigned short* qkvvT  = (unsigned short*) take((size_t)L * 4 * D * D * 2);
  unsigned short* WoT    = (unsigned short*) take((size_t)L * D * D * 2);
  unsigned short* W1T    = (unsigned short*) take((size_t)L * FF * D * 2);
  unsigned short* W2T    = (unsigned short*) take((size_t)L * D * FF * 2);
  unsigned short* qkvv   = (unsigned short*) take((size_t)M * 4 * D * 2);
  char*           sReg   =                   take((size_t)Bb * H * Nn * Nn * Nn * 4);
  unsigned short* oB     = (unsigned short*) take((size_t)M * D * 2);

  // Phase-aliased buffers (single-stream ordering makes this safe):
  float*          sBuf     = (float*)sReg;           // attn scores, then:
  float*          attnproj = (float*)sReg;           //   o@Wo output (16.8MB)
  unsigned short* ffnB     = (unsigned short*)sReg;  //   FFN hidden (33.6MB)
  float*          f2out    = (float*)qkvv;           // FFN out (qkvv dead by then)

  auto blk = [](long long threads) { return dim3((unsigned)((threads + 255) / 256)); };
  auto gblk = [](int Mr, int Nc) { return dim3((unsigned)((Mr / 128) * (Nc / 64))); };

  // --- weight / input conversion to bf16 (transposed, K zero-padded to 32) ---
  conv_pad_bf16<<<blk((long long)Bb * Nn * 32), 256, 0, stream>>>(x, Ax, Bb * Nn, 16, 32);
  conv_pad_bf16<<<blk((long long)M * 32), 256, 0, stream>>>(edge_attr, Ae, M, 8, 32);
  wconv_transpose_bf16<<<blk(D * 32), 256, 0, stream>>>(node_W, nodeWT, 16, D, 32);
  wconv_transpose_bf16<<<blk(D * 32), 256, 0, stream>>>(edge_W, edgeWT, 8, D, 32);
  const float* Wsp[4] = {Wq, Wk, Wv1, Wv2};
  for (int l = 0; l < L; ++l) {
    for (int p = 0; p < 4; ++p)
      wconv_transpose_bf16<<<blk(D * D), 256, 0, stream>>>(
          Wsp[p] + (size_t)l * D * D, qkvvT + ((size_t)l * 4 + p) * D * D, D, D, D);
    wconv_transpose_bf16<<<blk(D * D), 256, 0, stream>>>(
        Wo + (size_t)l * D * D, WoT + (size_t)l * D * D, D, D, D);
    wconv_transpose_bf16<<<blk(FF * D), 256, 0, stream>>>(
        W1 + (size_t)l * D * FF, W1T + (size_t)l * FF * D, D, FF, D);
    wconv_transpose_bf16<<<blk(D * FF), 256, 0, stream>>>(
        W2 + (size_t)l * FF * D, W2T + (size_t)l * D * FF, FF, D, FF);
  }

  // --- embeddings ---
  gemm_bf16_wmma_lds<<<gblk(M, D), 256, 0, stream>>>(
      Ae, 32, edgeWT, 32, tokF, (unsigned short*)nullptr, D, edge_b, M, D, 32, 0);
  gemm_bf16_wmma_lds<<<gblk(Bb * Nn, D), 256, 0, stream>>>(
      Ax, 32, nodeWT, 32, nodeEmb, (unsigned short*)nullptr, D, node_b, Bb * Nn, D, 32, 0);
  mask_embed<<<blk((long long)M * D), 256, 0, stream>>>(tokF, tokB, nodeEmb, no_edge, edge_mask);

  // --- layers ---
  for (int l = 0; l < L; ++l) {
    gemm_bf16_wmma_lds<<<gblk(M, 4 * D), 256, 0, stream>>>(
        tokB, D, qkvvT + (size_t)l * 4 * D * D, D,
        (float*)nullptr, qkvv, 4 * D, (const float*)nullptr, M, 4 * D, D, 0);
    attn_scores_wmma<<<blk((long long)Bb * H * Nn * 16 * 32), 256, 0, stream>>>(qkvv, sBuf);
    softmax_l<<<blk((long long)Bb * H * Nn * Nn), 256, 0, stream>>>(sBuf);
    tri_o_einsum<<<blk((long long)Bb * H * Nn * Nn * 4), 256, 0, stream>>>(sBuf, qkvv, oB);
    gemm_bf16_wmma_lds<<<gblk(M, D), 256, 0, stream>>>(
        oB, D, WoT + (size_t)l * D * D, D,
        attnproj, (unsigned short*)nullptr, D, bo + (size_t)l * D, M, D, D, 0);
    resid_layernorm<<<blk((long long)M * 32), 256, 0, stream>>>(
        tokF, tokB, attnproj, ln1g + (size_t)l * D, ln1b + (size_t)l * D);
    gemm_bf16_wmma_lds<<<gblk(M, FF), 256, 0, stream>>>(
        tokB, D, W1T + (size_t)l * FF * D, D,
        (float*)nullptr, ffnB, FF, b1 + (size_t)l * FF, M, FF, D, 1);
    gemm_bf16_wmma_lds<<<gblk(M, D), 256, 0, stream>>>(
        ffnB, FF, W2T + (size_t)l * D * FF, FF,
        f2out, (unsigned short*)nullptr, D, b2 + (size_t)l * D, M, D, FF, 0);
    resid_layernorm<<<blk((long long)M * 32), 256, 0, stream>>>(
        tokF, tokB, f2out, ln2g + (size_t)l * D, ln2b + (size_t)l * D);
  }

  head_out_k<<<blk((long long)Bb * Nn * 32), 256, 0, stream>>>(tokF, Wout, bout, out);
}
